// BasicLayer_3375844295247
// MI455X (gfx1250) — matrix operand order repaired and checked by reading, not compiled, not run on hardware
//
#include <hip/hip_runtime.h>
#include <hip/hip_bf16.h>
#include <math.h>

// ---------------- problem constants (match reference setup_inputs) ----------
constexpr int DEPTH = 2;
constexpr int Bq    = 4;
constexpr int GH    = 128;
constexpr int GW    = 128;
constexpr int Nq    = GH * GW;        // 16384 tokens per batch
constexpr int Dm    = 192;            // model dim
constexpr int NH    = 6;              // heads
constexpr int DH    = 32;             // head dim
constexpr int MCL   = 64;             // cluster size m
constexpr int KC    = Nq / MCL;       // clusters per batch
constexpr int DFF   = 4 * Dm;         // 768
constexpr long MTOK = (long)Bq * Nq;  // 65536 rows

typedef __attribute__((ext_vector_type(16))) __bf16 v16bf;
typedef __attribute__((ext_vector_type(8)))  float  v8f;
typedef __attribute__((ext_vector_type(8)))  __bf16 bf16x8;
typedef __attribute__((ext_vector_type(4)))  int    v4i;

// ---- gfx1250 async global->LDS copy (ASYNCcnt) with portable fallback ------
#if defined(__has_builtin)
#if __has_builtin(__builtin_amdgcn_global_load_async_to_lds_b128) && \
    __has_builtin(__builtin_amdgcn_s_wait_asynccnt)
#define USE_ASYNC_LDS 1
#endif
#endif
#ifndef USE_ASYNC_LDS
#define USE_ASYNC_LDS 0
#endif

#define AS1 __attribute__((address_space(1)))
#define AS3 __attribute__((address_space(3)))

static __device__ inline void copy16_to_lds(const __bf16* g, __bf16* l) {
#if USE_ASYNC_LDS
    __builtin_amdgcn_global_load_async_to_lds_b128((AS1 v4i*)(void*)g,
                                                   (AS3 v4i*)(void*)l, 0, 0);
#else
    *(bf16x8*)l = *(const bf16x8*)g;
#endif
}
static __device__ inline void copy_wait() {
#if USE_ASYNC_LDS
    __builtin_amdgcn_s_wait_asynccnt(0);
#endif
}

static __device__ inline float gelu_tanh(float x) {
    const float k0 = 0.7978845608028654f, k1 = 0.044715f;
    return 0.5f * x * (1.f + tanhf(k0 * (x + k1 * x * x * x)));
}

// ---------------------------------------------------------------------------
// inv_idx[b*Nq + j] = global ordered row (b*Nq + boustrophedon key) of token j
// ---------------------------------------------------------------------------
__global__ __launch_bounds__(256) void build_inv_idx(const float* __restrict__ pos,
                                                     int* __restrict__ inv) {
    long t = (long)blockIdx.x * 256 + threadIdx.x;
    if (t >= MTOK) return;
    long bi = t / Nq;
    int ix = (int)floorf(pos[t * 2 + 0]);
    int iy = (int)floorf(pos[t * 2 + 1]);
    int key = iy * GW + ((iy & 1) ? (GW - 1 - ix) : ix);
    inv[t] = (int)(bi * Nq + key);
}

// ---------------------------------------------------------------------------
// f32 -> bf16 bulk convert (weights), 4 elements / thread
// ---------------------------------------------------------------------------
__global__ __launch_bounds__(256) void cvt_f32_bf16(const float* __restrict__ src,
                                                    __bf16* __restrict__ dst, long n4) {
    long t = (long)blockIdx.x * 256 + threadIdx.x;
    if (t >= n4) return;
    float4 v = ((const float4*)src)[t];
    __bf16 o[4] = {(__bf16)v.x, (__bf16)v.y, (__bf16)v.z, (__bf16)v.w};
    ((ushort4*)dst)[t] = *(const ushort4*)o;
}

// ---------------------------------------------------------------------------
// LayerNorm (f32 stats) -> bf16 row.  One wave32 per token, 6 elems/lane.
// ---------------------------------------------------------------------------
__global__ __launch_bounds__(128) void layernorm_to_bf16(const float* __restrict__ x,
                                                         const float* __restrict__ g,
                                                         const float* __restrict__ b,
                                                         __bf16* __restrict__ out) {
    const int lane = threadIdx.x & 31, wave = threadIdx.x >> 5;
    const long tok = (long)blockIdx.x * 4 + wave;
    const float* row = x + tok * Dm;
    float vals[6], s = 0.f, ss = 0.f;
#pragma unroll
    for (int i = 0; i < 6; ++i) {
        float v = row[lane + 32 * i];
        vals[i] = v; s += v; ss += v * v;
    }
#pragma unroll
    for (int off = 16; off >= 1; off >>= 1) {
        s  += __shfl_xor(s,  off, 32);
        ss += __shfl_xor(ss, off, 32);
    }
    const float mu = s * (1.f / Dm);
    const float rs = rsqrtf(ss * (1.f / Dm) - mu * mu + 1e-5f);
#pragma unroll
    for (int i = 0; i < 6; ++i) {
        int c = lane + 32 * i;
        out[tok * Dm + c] = (__bf16)((vals[i] - mu) * rs * g[c] + b[c]);
    }
}

// ---------------------------------------------------------------------------
// WMMA fragment builders (wave32; A/B use this code's contiguous k order, not the ISA table order; C/D per ISA)
//   A 16x32 bf16: lane row = lane%16; kbase = (lane>>4)*16;
//                 vgpr j -> K = kbase+2j,+1
//   B 32x16 bf16: lane col = lane%16; kb = (lane>>4)*16; vgpr j -> K = kb+2j,+1
//   C/D f32 16x16: col = lane%16; vgpr r -> row = r + (lane>>4)*8
// ---------------------------------------------------------------------------
template <int STRIDE>
static __device__ inline v16bf load_a_frag(const __bf16 (*src)[STRIDE], int row0,
                                           int kOff, int lane) {
    v16bf a;
    const int r = row0 + (lane & 15);
    const int kb = kOff + (lane >> 4) * 16;
#pragma unroll
    for (int j = 0; j < 8; ++j) {
        a[2 * j]     = src[r][kb + 2 * j];
        a[2 * j + 1] = src[r][kb + 2 * j + 1];
    }
    return a;
}

// B fragment from LDS laid out [K][N]: element (kk, col) = src[kOff+kk][col0+col]
template <int STRIDE>
static __device__ inline v16bf load_b_frag(const __bf16 (*src)[STRIDE], int kOff,
                                           int col0, int lane) {
    v16bf b;
    const int c = col0 + (lane & 15);
    const int kb = kOff + (lane >> 4) * 16;
#pragma unroll
    for (int j = 0; j < 8; ++j) {
        b[2 * j]     = src[kb + 2 * j][c];
        b[2 * j + 1] = src[kb + 2 * j + 1][c];
    }
    return b;
}

// ---------------------------------------------------------------------------
// Generic bf16 GEMM: out(row,col) = A[row or idx[row]] @ W + bias (+resid)(gelu)
// Tile: 64x64, BK=32, 128 threads = 4 waves; wave owns a 32x32 quadrant.
// Double-buffered LDS tiles filled with async global->LDS copies (ASYNCcnt).
// ---------------------------------------------------------------------------
template <bool GATHER_A, bool SCATTER_OUT, bool ADD_RESID, bool DO_GELU, bool OUT_BF16>
__global__ __launch_bounds__(128) void gemm_bf16_wmma(
    const __bf16* __restrict__ A, const __bf16* __restrict__ W,
    const float* __restrict__ bias, const int* __restrict__ idx,
    const float* __restrict__ resid, float* __restrict__ outF,
    __bf16* __restrict__ outB, int N, int K) {
    __shared__ __bf16 a_s[2][64][40];   // 80B row stride, 16B aligned chunks
    __shared__ __bf16 b_s[2][32][72];   // 144B row stride
    const int tid = threadIdx.x, lane = tid & 31, wave = tid >> 5;
    const long rowBase = (long)blockIdx.x * 64;
    const int  colBase = blockIdx.y * 64;
    const int  wm = (wave >> 1) * 32;   // wave quadrant
    const int  wn = (wave & 1) * 32;

    auto load_tiles = [&](int k0, int buf) {
        // A tile 64x32: 256 16B chunks (4 per row), 2 per thread
#pragma unroll
        for (int i = tid; i < 256; i += 128) {
            int r = i >> 2, ch = (i & 3) * 8;
            long gr = rowBase + r;
            long ar = GATHER_A ? (long)idx[gr] : gr;
            copy16_to_lds(&A[ar * K + k0 + ch], &a_s[buf][r][ch]);
        }
        // B tile 32x64: 256 16B chunks (8 per row), 2 per thread
#pragma unroll
        for (int i = tid; i < 256; i += 128) {
            int r = i >> 3, ch = (i & 7) * 8;
            copy16_to_lds(&W[(long)(k0 + r) * N + colBase + ch], &b_s[buf][r][ch]);
        }
    };

    v8f acc[2][2] = {};
    const int nsteps = K / 32;
    load_tiles(0, 0);
    copy_wait();
    __syncthreads();
    for (int step = 0; step < nsteps; ++step) {
        const int cur = step & 1;
        if (step + 1 < nsteps) load_tiles((step + 1) * 32, cur ^ 1);

        v16bf af[2], bf[2];
#pragma unroll
        for (int mt = 0; mt < 2; ++mt)
            af[mt] = load_a_frag(a_s[cur], wm + mt * 16, 0, lane);
#pragma unroll
        for (int nt = 0; nt < 2; ++nt)
            bf[nt] = load_b_frag(b_s[cur], 0, wn + nt * 16, lane);
#pragma unroll
        for (int mt = 0; mt < 2; ++mt)
#pragma unroll
            for (int nt = 0; nt < 2; ++nt)
                acc[mt][nt] = __builtin_amdgcn_wmma_f32_16x16x32_bf16(
                    false, af[mt], false, bf[nt], (short)0, acc[mt][nt], false, false);

        copy_wait();       // next tile's async fills complete
        __syncthreads();   // all waves done reading cur, next buffer visible
    }
    // --- epilogue ---
#pragma unroll
    for (int mt = 0; mt < 2; ++mt) {
#pragma unroll
        for (int nt = 0; nt < 2; ++nt) {
            const int col = colBase + wn + nt * 16 + (lane & 15);
            const float bv = bias ? bias[col] : 0.f;
#pragma unroll
            for (int r = 0; r < 8; ++r) {
                long grow = rowBase + wm + mt * 16 + ((lane >> 4) << 3) + r;
                float v = acc[mt][nt][r] + bv;
                if (ADD_RESID) v += resid[grow * N + col];
                if (DO_GELU) v = gelu_tanh(v);
                long orow = SCATTER_OUT ? (long)idx[grow] : grow;
                if (OUT_BF16) outB[orow * N + col] = (__bf16)v;
                else          outF[orow * N + col] = v;
            }
        }
    }
}

// ---------------------------------------------------------------------------
// Cluster attention: one block = (batch*cluster, head), 128 threads = 4 waves.
// qkv is curve-ordered bf16 (row, 576) with col layout [t(3)][head(6)][dh(32)].
// S = Q K^T (64x64, f32 accum), softmax f32, O = P V (64x32) -> ordered bf16 o.
// ---------------------------------------------------------------------------
__global__ __launch_bounds__(128) void attn_cluster(const __bf16* __restrict__ qkv,
                                                    __bf16* __restrict__ o) {
    const int head = blockIdx.y;
    const long base = (long)blockIdx.x * MCL;  // ordered row of cluster start
    const int tid = threadIdx.x, lane = tid & 31, wave = tid >> 5;

    __shared__ __bf16 q_s[64][40];
    __shared__ __bf16 k_s[64][40];
    __shared__ __bf16 v_s[64][40];
    __shared__ float  s_s[64][68];   // 272B row stride (16B aligned)
    __shared__ __bf16 p_s[64][72];

    const int qoff = head * DH, koff = Dm + head * DH, voff = 2 * Dm + head * DH;
    // 64 rows x 4 chunks of 8 bf16 per matrix; 2 chunks/thread/matrix (async)
#pragma unroll
    for (int i = tid; i < 256; i += 128) {
        int r = i >> 2, ch = (i & 3) * 8;
        long rb = (base + r) * (3 * Dm);
        copy16_to_lds(&qkv[rb + qoff + ch], &q_s[r][ch]);
        copy16_to_lds(&qkv[rb + koff + ch], &k_s[r][ch]);
        copy16_to_lds(&qkv[rb + voff + ch], &v_s[r][ch]);
    }
    copy_wait();
    __syncthreads();

    // --- S = Q K^T : wave w computes rows [16w,16w+16) against all 64 cols ---
    {
        const v16bf a = load_a_frag(q_s, wave * 16, 0, lane);
#pragma unroll
        for (int nt = 0; nt < 4; ++nt) {
            // B = K^T : element (kk, col) = K[col][kk] -> use A-style row access
            v16bf bb;
            const int colr = nt * 16 + (lane & 15);
            const int kb = (lane >> 4) * 16;
#pragma unroll
            for (int j = 0; j < 8; ++j) {
                bb[2 * j]     = k_s[colr][kb + 2 * j];
                bb[2 * j + 1] = k_s[colr][kb + 2 * j + 1];
            }
            v8f c = {};
            c = __builtin_amdgcn_wmma_f32_16x16x32_bf16(
                false, a, false, bb, (short)0, c, false, false);
            const int c0 = nt * 16 + (lane & 15);
            const int r0 = wave * 16 + ((lane >> 4) << 3);
#pragma unroll
            for (int r = 0; r < 8; ++r) s_s[r0 + r][c0] = c[r];
        }
    }
    __syncthreads();

    // --- softmax (f32): 2 threads per row (partner lanes tid^1), float4 reads ---
    {
        const float scale = 0.17677669529663687f;  // 1/sqrt(32)
        const int row = tid >> 1;
        const int c0 = (tid & 1) * 32;
        float4 chunk[8];
        float mx = -3.0e38f;
#pragma unroll
        for (int i = 0; i < 8; ++i) {
            float4 v = *(const float4*)(&s_s[row][c0 + 4 * i]);
            v.x *= scale; v.y *= scale; v.z *= scale; v.w *= scale;
            chunk[i] = v;
            mx = fmaxf(mx, fmaxf(fmaxf(v.x, v.y), fmaxf(v.z, v.w)));
        }
        mx = fmaxf(mx, __shfl_xor(mx, 1, 32));   // combine row halves (same wave)
        float sum = 0.f;
#pragma unroll
        for (int i = 0; i < 8; ++i) {
            chunk[i].x = __expf(chunk[i].x - mx);
            chunk[i].y = __expf(chunk[i].y - mx);
            chunk[i].z = __expf(chunk[i].z - mx);
            chunk[i].w = __expf(chunk[i].w - mx);
            sum += chunk[i].x + chunk[i].y + chunk[i].z + chunk[i].w;
        }
        sum += __shfl_xor(sum, 1, 32);
        const float inv = 1.f / sum;
#pragma unroll
        for (int i = 0; i < 8; ++i) {
            p_s[row][c0 + 4 * i + 0] = (__bf16)(chunk[i].x * inv);
            p_s[row][c0 + 4 * i + 1] = (__bf16)(chunk[i].y * inv);
            p_s[row][c0 + 4 * i + 2] = (__bf16)(chunk[i].z * inv);
            p_s[row][c0 + 4 * i + 3] = (__bf16)(chunk[i].w * inv);
        }
    }
    __syncthreads();

    // --- O = P V : wave w rows [16w,16w+16), 2 n-tiles, K=64 in 2 steps ---
    {
        const int r0 = wave * 16;
#pragma unroll
        for (int nt = 0; nt < 2; ++nt) {
            v8f c = {};
#pragma unroll
            for (int s = 0; s < 2; ++s) {
                const v16bf a = load_a_frag(p_s, r0, s * 32, lane);
                const v16bf bb = load_b_frag(v_s, s * 32, nt * 16, lane);
                c = __builtin_amdgcn_wmma_f32_16x16x32_bf16(
                    false, a, false, bb, (short)0, c, false, false);
            }
            const int gcol = head * DH + nt * 16 + (lane & 15);
#pragma unroll
            for (int r = 0; r < 8; ++r) {
                long grow = base + r0 + ((lane >> 4) << 3) + r;
                o[grow * Dm + gcol] = (__bf16)c[r];
            }
        }
    }
}

// ---------------------------------------------------------------------------
extern "C" void kernel_launch(void* const* d_in, const int* in_sizes, int n_in,
                              void* d_out, int out_size, void* d_ws, size_t ws_size,
                              hipStream_t stream) {
    const float* x     = (const float*)d_in[0];
    const float* pos   = (const float*)d_in[1];
    const float* ln1_g = (const float*)d_in[2];
    const float* ln1_b = (const float*)d_in[3];
    const float* wqkv  = (const float*)d_in[4];
    const float* bqkv  = (const float*)d_in[5];
    const float* wproj = (const float*)d_in[6];
    const float* bproj = (const float*)d_in[7];
    const float* ln2_g = (const float*)d_in[8];
    const float* ln2_b = (const float*)d_in[9];
    const float* wfc1  = (const float*)d_in[10];
    const float* bfc1  = (const float*)d_in[11];
    const float* wfc2  = (const float*)d_in[12];
    const float* bfc2  = (const float*)d_in[13];

    float* xbuf = (float*)d_out;  // working x lives in d_out

    // workspace carve
    char* ws = (char*)d_ws;
    size_t off = 0;
    auto alloc = [&](size_t bytes) -> void* {
        void* p = ws + off;
        off += (bytes + 255) & ~(size_t)255;
        return p;
    };
    __bf16* xn   = (__bf16*)alloc(MTOK * Dm * sizeof(__bf16));
    __bf16* big  = (__bf16*)alloc(MTOK * DFF * sizeof(__bf16));  // qkv(576) / mlp h(768)
    __bf16* obuf = (__bf16*)alloc(MTOK * Dm * sizeof(__bf16));
    int*    inv  = (int*)alloc(MTOK * sizeof(int));
    __bf16* wqkv_bf = (__bf16*)alloc((size_t)DEPTH * Dm * 3 * Dm * sizeof(__bf16));
    __bf16* wproj_bf = (__bf16*)alloc((size_t)DEPTH * Dm * Dm * sizeof(__bf16));
    __bf16* wfc1_bf = (__bf16*)alloc((size_t)DEPTH * Dm * DFF * sizeof(__bf16));
    __bf16* wfc2_bf = (__bf16*)alloc((size_t)DEPTH * DFF * Dm * sizeof(__bf16));

    (void)hipMemcpyAsync(xbuf, x, (size_t)MTOK * Dm * sizeof(float),
                         hipMemcpyDeviceToDevice, stream);

    build_inv_idx<<<(int)((MTOK + 255) / 256), 256, 0, stream>>>(pos, inv);

    auto cvt = [&](const float* s, __bf16* d, long n) {
        long n4 = n / 4;
        cvt_f32_bf16<<<(int)((n4 + 255) / 256), 256, 0, stream>>>(s, d, n4);
    };
    cvt(wqkv,  wqkv_bf,  (long)DEPTH * Dm * 3 * Dm);
    cvt(wproj, wproj_bf, (long)DEPTH * Dm * Dm);
    cvt(wfc1,  wfc1_bf,  (long)DEPTH * Dm * DFF);
    cvt(wfc2,  wfc2_bf,  (long)DEPTH * DFF * Dm);

    const int MT = (int)(MTOK / 64);  // 1024 row tiles
    for (int l = 0; l < DEPTH; ++l) {
        // LN1 -> xn
        layernorm_to_bf16<<<(int)(MTOK / 4), 128, 0, stream>>>(
            xbuf, ln1_g + l * Dm, ln1_b + l * Dm, xn);
        // qkv = xn @ Wqkv + b, scattered into curve order (bf16)
        gemm_bf16_wmma<false, true, false, false, true>
            <<<dim3(MT, 3 * Dm / 64), 128, 0, stream>>>(
                xn, wqkv_bf + (size_t)l * Dm * 3 * Dm, bqkv + l * 3 * Dm, inv,
                nullptr, nullptr, big, 3 * Dm, Dm);
        // clustered attention -> obuf (ordered bf16)
        attn_cluster<<<dim3(Bq * KC, NH), 128, 0, stream>>>(big, obuf);
        // x += gather(o) @ Wproj + b
        gemm_bf16_wmma<true, false, true, false, false>
            <<<dim3(MT, Dm / 64), 128, 0, stream>>>(
                obuf, wproj_bf + (size_t)l * Dm * Dm, bproj + l * Dm, inv,
                xbuf, xbuf, nullptr, Dm, Dm);
        // LN2 -> xn
        layernorm_to_bf16<<<(int)(MTOK / 4), 128, 0, stream>>>(
            xbuf, ln2_g + l * Dm, ln2_b + l * Dm, xn);
        // h = gelu(xn @ Wfc1 + b)  (bf16)
        gemm_bf16_wmma<false, false, false, true, true>
            <<<dim3(MT, DFF / 64), 128, 0, stream>>>(
                xn, wfc1_bf + (size_t)l * Dm * DFF, bfc1 + l * DFF, nullptr,
                nullptr, nullptr, big, DFF, Dm);
        // x += h @ Wfc2 + b
        gemm_bf16_wmma<false, false, true, false, false>
            <<<dim3(MT, Dm / 64), 128, 0, stream>>>(
                big, wfc2_bf + (size_t)l * DFF * Dm, bfc2 + l * Dm, nullptr,
                xbuf, xbuf, nullptr, Dm, DFF);
    }
}
